// MacroEncoder_18442589569695
// MI455X (gfx1250) — matrix operand order repaired and checked by reading, not compiled, not run on hardware
//
#include <hip/hip_runtime.h>
#include <hip/hip_bf16.h>

// ---------------- problem constants ----------------
#define L_SEQ     16384
#define DMODEL    256
#define DINNER    512
#define DSTATE    64
#define NHEADS    8
#define HEADDIM   64
#define CONVDIM   640
#define DPROJ     1160
#define DPROJ_PAD 1216   // 19 * 64, padded so GEMM-N has no guards
#define NLAYERS   4

typedef __attribute__((ext_vector_type(16))) _Float16 v16h;
typedef __attribute__((ext_vector_type(8)))  float    v8f;
typedef __attribute__((ext_vector_type(4)))  int      v4i_t;

union Frag { unsigned u[8]; v16h h; };

// ---- CDNA5 async global->LDS path (guarded; falls back to sync staging) ----
// Probe-learned signatures: b128 takes (as1 int4*, as3 int4*, imm, imm),
// b32 takes (as1 int*, as3 int*, imm, imm). addrspace(1) prints as __device__.
#if defined(__HIP_DEVICE_COMPILE__) && defined(__gfx1250__) &&                 \
    __has_builtin(__builtin_amdgcn_global_load_async_to_lds_b128) &&           \
    __has_builtin(__builtin_amdgcn_global_load_async_to_lds_b32) &&            \
    __has_builtin(__builtin_amdgcn_s_wait_asynccnt)
#define USE_ASYNC 1
typedef __attribute__((address_space(1))) v4i_t* g128p;
typedef __attribute__((address_space(3))) v4i_t* l128p;
typedef __attribute__((address_space(1))) int*   g32p;
typedef __attribute__((address_space(3))) int*   l32p;
#define ASYNC_B128(g, l)                                                       \
    __builtin_amdgcn_global_load_async_to_lds_b128((g128p)(void*)(g),          \
                                                   (l128p)(void*)(l), 0, 0)
#define ASYNC_B32(g, l)                                                        \
    __builtin_amdgcn_global_load_async_to_lds_b32((g32p)(void*)(g),            \
                                                  (l32p)(void*)(l), 0, 0)
#define ASYNC_WAIT() __builtin_amdgcn_s_wait_asynccnt(0)
#else
#define USE_ASYNC 0
#define ASYNC_WAIT()
#endif

__device__ __forceinline__ float silu_f(float x) { return x / (1.f + expf(-x)); }

// ---------------- f32 -> f16 weight conversion with N padding ----------------
__global__ void convert_pad_f16(const float* __restrict__ src, _Float16* __restrict__ dst,
                                int K, int N, int Npad) {
    int idx = blockIdx.x * blockDim.x + threadIdx.x;
    if (idx >= K * Npad) return;
    int k = idx / Npad, n = idx - k * Npad;
    dst[idx] = (n < N) ? (_Float16)src[(size_t)k * N + n] : (_Float16)0.f;
}

// ---------------- input projection: x = tokens(16384x10) @ W(10x256) + b ----------------
__global__ void input_proj_kernel(const float* __restrict__ tok, const float* __restrict__ W,
                                  const float* __restrict__ b, float* __restrict__ x) {
    int idx = blockIdx.x * blockDim.x + threadIdx.x;   // t*256 + d
    int t = idx >> 8, d = idx & 255;
    float acc = b[d];
#pragma unroll
    for (int k = 0; k < 10; ++k) acc += tok[t * 10 + k] * W[k * 256 + d];
    x[idx] = acc;
}

// ---------------- rmsnorm over 256 + cast to f16 ----------------
__global__ __launch_bounds__(256) void rmsnorm_f16_kernel(const float* __restrict__ x,
                                                          const float* __restrict__ w,
                                                          _Float16* __restrict__ xn) {
    int t = blockIdx.x;
    __shared__ float red[256];
    float v = x[(size_t)t * 256 + threadIdx.x];
    red[threadIdx.x] = v * v;
    __syncthreads();
    for (int s = 128; s > 0; s >>= 1) {
        if (threadIdx.x < s) red[threadIdx.x] += red[threadIdx.x + s];
        __syncthreads();
    }
    float scale = rsqrtf(red[0] * (1.f / 256.f) + 1e-5f);
    xn[(size_t)t * 256 + threadIdx.x] = (_Float16)(v * scale * w[threadIdx.x]);
}

// ---------------- WMMA GEMM: C(MxN,f32) [+]= A(MxK,f16) @ B(KxN,f16) + bias ----------------
// block tile 128x64, 256 threads (8 waves), wave w -> rows [16w,16w+16), 4 wmma n-tiles
template <bool ACCUM>
__global__ __launch_bounds__(256) void gemm_f16_wmma(const _Float16* __restrict__ A, int lda,
                                                     const _Float16* __restrict__ B, int ldb,
                                                     float* __restrict__ C, int ldc,
                                                     int K, int N, const float* __restrict__ bias) {
    __shared__ _Float16 As[128][32];
    __shared__ _Float16 BsT[64][32];   // n-major so K pairs are contiguous

    const int tid  = threadIdx.x;
    const int wave = tid >> 5;
    const int lane = tid & 31;
    const int m0 = blockIdx.x * 128;
    const int n0 = blockIdx.y * 64;

    v8f acc[4];
#pragma unroll
    for (int i = 0; i < 4; ++i)
#pragma unroll
        for (int e = 0; e < 8; ++e) acc[i][e] = 0.f;

    const int aRow = tid >> 1;
    const int aCol = (tid & 1) * 16;
    const int bN   = tid & 63;
    const int bK0  = (tid >> 6) * 8;

    for (int k0 = 0; k0 < K; k0 += 32) {
        const _Float16* ag = A + (size_t)(m0 + aRow) * lda + k0 + aCol;
#if USE_ASYNC
        // async DMA straight into LDS: no VGPR round-trip, overlaps with B staging
        ASYNC_B128(ag, &As[aRow][aCol]);
        ASYNC_B128(ag + 8, &As[aRow][aCol + 8]);
#else
        *(uint4*)&As[aRow][aCol]     = *(const uint4*)ag;
        *(uint4*)&As[aRow][aCol + 8] = *(const uint4*)(ag + 8);
#endif
        // B staged transposed (n-major) so K-pairs are contiguous for fragments
#pragma unroll
        for (int j = 0; j < 8; ++j)
            BsT[bN][bK0 + j] = B[(size_t)(k0 + bK0 + j) * ldb + n0 + bN];
        ASYNC_WAIT();
        __syncthreads();

        const int hl = lane >> 4;
        const int ml = lane & 15;
        Frag a;
#pragma unroll
        for (int v = 0; v < 8; ++v) {
            int k = ((v < 4) ? 2 * v : 8 + 2 * v) + hl * 8;   // A layout per ISA
            a.u[v] = *(const unsigned*)&As[wave * 16 + ml][k];
        }
#pragma unroll
        for (int nt = 0; nt < 4; ++nt) {
            Frag b;
#pragma unroll
            for (int v = 0; v < 8; ++v) {
                int k = ((v < 4) ? 2 * v : 8 + 2 * v) + hl * 8;   // B layout matches A (table order)
                b.u[v] = *(const unsigned*)&BsT[nt * 16 + ml][k];
            }
            acc[nt] = __builtin_amdgcn_wmma_f32_16x16x32_f16(
                false, a.h, false, b.h, (short)0, acc[nt], false, false);
        }
        __syncthreads();
    }

    const int hl = lane >> 4, nl = lane & 15;
#pragma unroll
    for (int nt = 0; nt < 4; ++nt) {
        int n = n0 + nt * 16 + nl;
        float bi = (bias != nullptr && n < N) ? bias[n] : 0.f;
#pragma unroll
        for (int r = 0; r < 8; ++r) {
            int m = m0 + wave * 16 + r + hl * 8;
            size_t idx = (size_t)m * ldc + n;
            float v = acc[nt][r] + bi;
            if (ACCUM) v += C[idx];
            C[idx] = v;
        }
    }
}

// ---------------- depthwise causal conv (dir-aware) + bias + silu ----------------
// dir 0: taps t-3..t with w[0..3]; dir 1 (time-reversed causal): taps t+3..t with w[0..3]
__global__ void conv_silu_kernel(const float* __restrict__ zx, const float* __restrict__ cw,
                                 const float* __restrict__ cb, float* __restrict__ out, int dir) {
    int idx = blockIdx.x * blockDim.x + threadIdx.x;   // t*640 + c
    if (idx >= L_SEQ * CONVDIM) return;
    int t = idx / CONVDIM, c = idx - t * CONVDIM;
    float acc = cb[c];
#pragma unroll
    for (int j = 0; j < 4; ++j) {
        int tt = dir ? (t + 3 - j) : (t - 3 + j);
        if (tt >= 0 && tt < L_SEQ)
            acc += cw[c * 4 + j] * zx[(size_t)tt * DPROJ_PAD + DINNER + c];
    }
    out[idx] = silu_f(acc);
}

// ---------------- dt = softplus(raw + bias), dA = exp(-exp(Alog)*dt) ----------------
__global__ void dt_prep_kernel(const float* __restrict__ zx, const float* __restrict__ dtb,
                               const float* __restrict__ Alog, float* __restrict__ dtv,
                               float* __restrict__ dAv) {
    int idx = blockIdx.x * blockDim.x + threadIdx.x;   // t*8 + h
    if (idx >= L_SEQ * NHEADS) return;
    int t = idx >> 3, h = idx & 7;
    float v = zx[(size_t)t * DPROJ_PAD + (DINNER + CONVDIM) + h] + dtb[h];
    float dt = (v > 20.f) ? v : log1pf(expf(v));
    dtv[idx] = dt;
    dAv[idx] = expf(-expf(Alog[h]) * dt);
}

// ---------------- sequential selective scan: one block per head ----------------
// 256 threads: p = tid>>2 (0..63), nseg = tid&3 covering n in [16*nseg,16*nseg+16)
#define SCHUNK 64
__global__ __launch_bounds__(256) void scan_kernel(const float* __restrict__ conv,
                                                   const float* __restrict__ dtv,
                                                   const float* __restrict__ dAv,
                                                   float* __restrict__ ys, int dir) {
    const int h = blockIdx.x;
    const int tid = threadIdx.x;
    const int p = tid >> 2;
    const int nseg = tid & 3;
    const int nbase = nseg * 16;

    __shared__ float Bs[SCHUNK][DSTATE];
    __shared__ float Cs[SCHUNK][DSTATE];
    __shared__ float Xs[SCHUNK][HEADDIM];
    __shared__ float dts[SCHUNK];
    __shared__ float dAs[SCHUNK];

    float hs[16];
#pragma unroll
    for (int n = 0; n < 16; ++n) hs[n] = 0.f;

    const int g = tid >> 6;        // 0..3 : step group for cooperative loads
    const int c = tid & 63;        // column

    for (int chunk = 0; chunk < L_SEQ / SCHUNK; ++chunk) {
        int base = chunk * SCHUNK;
#pragma unroll 4
        for (int r = 0; r < 16; ++r) {
            int i = g * 16 + r;
            int t = dir ? (L_SEQ - 1 - (base + i)) : (base + i);
            size_t row = (size_t)t * CONVDIM;
#if USE_ASYNC
            ASYNC_B32(&conv[row + DINNER + c],          &Bs[i][c]);
            ASYNC_B32(&conv[row + DINNER + DSTATE + c], &Cs[i][c]);
            ASYNC_B32(&conv[row + h * HEADDIM + c],     &Xs[i][c]);
#else
            Bs[i][c] = conv[row + DINNER + c];
            Cs[i][c] = conv[row + DINNER + DSTATE + c];
            Xs[i][c] = conv[row + h * HEADDIM + c];
#endif
        }
        if (tid < SCHUNK) {
            int t = dir ? (L_SEQ - 1 - (base + tid)) : (base + tid);
#if USE_ASYNC
            ASYNC_B32(&dtv[(size_t)t * NHEADS + h], &dts[tid]);
            ASYNC_B32(&dAv[(size_t)t * NHEADS + h], &dAs[tid]);
#else
            dts[tid] = dtv[(size_t)t * NHEADS + h];
            dAs[tid] = dAv[(size_t)t * NHEADS + h];
#endif
        }
        ASYNC_WAIT();
        __syncthreads();

        for (int i = 0; i < SCHUNK; ++i) {
            float dA = dAs[i];
            float coef = dts[i] * Xs[i][p];
            float acc = 0.f;
#pragma unroll
            for (int n = 0; n < 16; ++n) {
                hs[n] = fmaf(hs[n], dA, coef * Bs[i][nbase + n]);
                acc = fmaf(hs[n], Cs[i][nbase + n], acc);
            }
            // reduce across the 4 lanes sharing this p (lanes l, l^1, l^2 in-wave)
            acc += __shfl_xor(acc, 1, 32);
            acc += __shfl_xor(acc, 2, 32);
            if (nseg == 0) {
                int t = dir ? (L_SEQ - 1 - (base + i)) : (base + i);
                ys[(size_t)t * DINNER + h * HEADDIM + p] = acc;
            }
        }
        __syncthreads();
    }
}

// ---------------- gated rmsnorm over 512, cast to f16 ----------------
__global__ __launch_bounds__(256) void gate_norm_kernel(const float* __restrict__ ys,
                                                        const float* __restrict__ conv,
                                                        const float* __restrict__ zx,
                                                        const float* __restrict__ Dh,
                                                        const float* __restrict__ gw,
                                                        _Float16* __restrict__ y16) {
    int t = blockIdx.x;
    __shared__ float red[256];
    float g[2];
    float ss = 0.f;
#pragma unroll
    for (int j = 0; j < 2; ++j) {
        int i = threadIdx.x + j * 256;
        float xs = conv[(size_t)t * CONVDIM + i];
        float yv = ys[(size_t)t * DINNER + i] + Dh[i >> 6] * xs;
        float z = zx[(size_t)t * DPROJ_PAD + i];
        float gg = yv * silu_f(z);
        g[j] = gg;
        ss += gg * gg;
    }
    red[threadIdx.x] = ss;
    __syncthreads();
    for (int s = 128; s > 0; s >>= 1) {
        if (threadIdx.x < s) red[threadIdx.x] += red[threadIdx.x + s];
        __syncthreads();
    }
    float scale = rsqrtf(red[0] * (1.f / 512.f) + 1e-5f);
#pragma unroll
    for (int j = 0; j < 2; ++j) {
        int i = threadIdx.x + j * 256;
        y16[(size_t)t * DINNER + i] = (_Float16)(g[j] * scale * gw[i]);
    }
}

// ---------------- column mean over 16384 rows ----------------
__global__ __launch_bounds__(256) void colmean_kernel(const float* __restrict__ x,
                                                      float* __restrict__ mv) {
    int d = blockIdx.x;
    __shared__ float red[256];
    float s = 0.f;
    for (int r = threadIdx.x; r < L_SEQ; r += 256) s += x[(size_t)r * 256 + d];
    red[threadIdx.x] = s;
    __syncthreads();
    for (int st = 128; st > 0; st >>= 1) {
        if (threadIdx.x < st) red[threadIdx.x] += red[threadIdx.x + st];
        __syncthreads();
    }
    if (threadIdx.x == 0) mv[d] = red[0] * (1.f / (float)L_SEQ);
}

// ---------------- Z = mean @ W_z + b_z ----------------
__global__ void final_kernel(const float* __restrict__ mv, const float* __restrict__ Wz,
                             const float* __restrict__ bz, float* __restrict__ out) {
    int n = threadIdx.x;
    float acc = bz[n];
    for (int k = 0; k < 256; ++k) acc += mv[k] * Wz[k * 256 + n];
    out[n] = acc;
}

// ---------------- host orchestration ----------------
extern "C" void kernel_launch(void* const* d_in, const int* in_sizes, int n_in,
                              void* d_out, int out_size, void* d_ws, size_t ws_size,
                              hipStream_t stream) {
    const float* tokens  = (const float*)d_in[0];
    const float* W_inp   = (const float*)d_in[1];
    const float* b_inp   = (const float*)d_in[2];
    const float* ln_w    = (const float*)d_in[3];
    const float* W_in    = (const float*)d_in[4];
    const float* b_in    = (const float*)d_in[5];
    const float* conv_w  = (const float*)d_in[6];
    const float* conv_b  = (const float*)d_in[7];
    const float* dt_bias = (const float*)d_in[8];
    const float* A_log   = (const float*)d_in[9];
    const float* D_par   = (const float*)d_in[10];
    const float* gnorm_w = (const float*)d_in[11];
    const float* W_out   = (const float*)d_in[12];
    const float* b_out   = (const float*)d_in[13];
    const float* W_z     = (const float*)d_in[14];
    const float* b_z     = (const float*)d_in[15];

    char* p = (char*)d_ws;
    auto alloc = [&](size_t b) -> char* {
        char* r = p;
        p += (b + 255) & ~(size_t)255;
        return r;
    };
    float*    xf      = (float*)alloc((size_t)L_SEQ * 256 * 4);
    _Float16* xn16    = (_Float16*)alloc((size_t)L_SEQ * 256 * 2);
    float*    zx      = (float*)alloc((size_t)L_SEQ * DPROJ_PAD * 4);
    float*    conv    = (float*)alloc((size_t)L_SEQ * CONVDIM * 4);
    float*    dtv     = (float*)alloc((size_t)L_SEQ * NHEADS * 4);
    float*    dAv     = (float*)alloc((size_t)L_SEQ * NHEADS * 4);
    float*    ysb     = (float*)alloc((size_t)L_SEQ * DINNER * 4);
    _Float16* y16     = (_Float16*)alloc((size_t)L_SEQ * DINNER * 2);
    _Float16* w_in16  = (_Float16*)alloc((size_t)8 * 256 * DPROJ_PAD * 2);
    _Float16* w_out16 = (_Float16*)alloc((size_t)8 * 512 * 256 * 2);
    float*    meanv   = (float*)alloc(256 * 4);

    // weight conversion (deterministic, every call)
    for (int ld = 0; ld < 8; ++ld) {
        convert_pad_f16<<<(256 * DPROJ_PAD + 255) / 256, 256, 0, stream>>>(
            W_in + (size_t)ld * 256 * DPROJ, w_in16 + (size_t)ld * 256 * DPROJ_PAD,
            256, DPROJ, DPROJ_PAD);
        convert_pad_f16<<<(512 * 256 + 255) / 256, 256, 0, stream>>>(
            W_out + (size_t)ld * 512 * 256, w_out16 + (size_t)ld * 512 * 256,
            512, 256, 256);
    }

    input_proj_kernel<<<(L_SEQ * 256) / 256, 256, 0, stream>>>(tokens, W_inp, b_inp, xf);

    for (int l = 0; l < NLAYERS; ++l) {
        rmsnorm_f16_kernel<<<L_SEQ, 256, 0, stream>>>(xf, ln_w + l * 256, xn16);
        for (int dir = 0; dir < 2; ++dir) {
            int ld = l * 2 + dir;
            // in-projection (pointwise in t: direction handled downstream)
            gemm_f16_wmma<false><<<dim3(L_SEQ / 128, DPROJ_PAD / 64), 256, 0, stream>>>(
                xn16, 256, w_in16 + (size_t)ld * 256 * DPROJ_PAD, DPROJ_PAD,
                zx, DPROJ_PAD, 256, DPROJ, b_in + (size_t)ld * DPROJ);
            conv_silu_kernel<<<(L_SEQ * CONVDIM + 255) / 256, 256, 0, stream>>>(
                zx, conv_w + (size_t)ld * CONVDIM * 4, conv_b + (size_t)ld * CONVDIM,
                conv, dir);
            dt_prep_kernel<<<(L_SEQ * NHEADS + 255) / 256, 256, 0, stream>>>(
                zx, dt_bias + ld * NHEADS, A_log + ld * NHEADS, dtv, dAv);
            scan_kernel<<<NHEADS, 256, 0, stream>>>(conv, dtv, dAv, ysb, dir);
            gate_norm_kernel<<<L_SEQ, 256, 0, stream>>>(
                ysb, conv, zx, D_par + ld * NHEADS, gnorm_w + (size_t)ld * DINNER, y16);
            // out-projection accumulates into the residual stream
            gemm_f16_wmma<true><<<dim3(L_SEQ / 128, 256 / 64), 256, 0, stream>>>(
                y16, 512, w_out16 + (size_t)ld * 512 * 256, 256,
                xf, 256, 512, 256, b_out + (size_t)ld * 256);
        }
    }

    colmean_kernel<<<256, 256, 0, stream>>>(xf, meanv);
    final_kernel<<<1, 256, 0, stream>>>(meanv, W_z, b_z, (float*)d_out);
}
